// DataEmbedding_67817533604482
// MI455X (gfx1250) — matrix operand order repaired and checked by reading, not compiled, not run on hardware
//
#include <hip/hip_runtime.h>

typedef __attribute__((ext_vector_type(16))) _Float16 v16h;
typedef __attribute__((ext_vector_type(8)))  _Float16 v8h;
typedef __attribute__((ext_vector_type(8)))  float    v8f;

#define BATCH   32
#define SEQ     512
#define CIN     512
#define DMODEL  1024

#define BM 128
#define BN 128
#define BK 32
#define LDSROW 40   // padded row stride in halves: n*80B -> 16 distinct banks per 16 lanes

// ---------------------------------------------------------------------------
// Kernel 1: write prepended subject-embedding row (out[b, 0, :])
// ---------------------------------------------------------------------------
__global__ __launch_bounds__(256)
void embed_rows_kernel(const float* __restrict__ subj_table,
                       const int*   __restrict__ sids,
                       float*       __restrict__ out) {
    const int b = blockIdx.x;
    const int sid = sids[b];
    const float* src = subj_table + (size_t)sid * DMODEL;
    float* dst = out + (size_t)b * (SEQ + 1) * DMODEL;
    for (int d = threadIdx.x; d < DMODEL; d += 256)
        dst[d] = src[d];
}

// ---------------------------------------------------------------------------
// Kernel 2: per-subject GEMM + bias + mask blend, written to out[b, 1+s, :]
// Split-f16 WMMA: x = xh + xl, w = wh + wl; acc += xh*wh + xh*wl + xl*wh
// ---------------------------------------------------------------------------
__global__ __launch_bounds__(256)
void subject_gemm_kernel(const float* __restrict__ x,
                         const float* __restrict__ mask,
                         const float* __restrict__ W,
                         const float* __restrict__ bias,
                         const float* __restrict__ mask_token,
                         const int*   __restrict__ sids,
                         float*       __restrict__ out) {
    __shared__ _Float16 Ah[BM * LDSROW];
    __shared__ _Float16 Al[BM * LDSROW];
    __shared__ _Float16 Bh[BN * LDSROW];  // transposed: [n][k]
    __shared__ _Float16 Bl[BN * LDSROW];

    const int t       = threadIdx.x;
    const int lane    = t & 31;
    const int wave    = t >> 5;
    const int wm      = (wave & 3) * 32;   // 4 waves along M (4*32 = 128)
    const int wn      = (wave >> 2) * 64;  // 2 waves along N (2*64 = 128)
    const int lrow    = lane & 15;
    const int halfsel = lane >> 4;

    const int s0 = blockIdx.x * BM;
    const int n0 = blockIdx.y * BN;
    const int b  = blockIdx.z;
    const int sid = sids[b];

    const float* xb = x + ((size_t)b * SEQ + s0) * CIN;
    const float* Wb = W + (size_t)sid * CIN * DMODEL + n0;

    v8f acc[2][4];
    #pragma unroll
    for (int mi = 0; mi < 2; ++mi)
        #pragma unroll
        for (int ni = 0; ni < 4; ++ni)
            acc[mi][ni] = (v8f){0.f, 0.f, 0.f, 0.f, 0.f, 0.f, 0.f, 0.f};

    for (int kt = 0; kt < CIN; kt += BK) {
        // ---- stage A tile (128 x 32 fp32 -> f16 hi/lo) ----
        #pragma unroll
        for (int i = 0; i < 4; ++i) {
            const int idx = t + i * 256;            // 1024 float4s
            const int row = idx >> 3;
            const int c4  = (idx & 7) << 2;
            const float4 v = *(const float4*)(xb + (size_t)row * CIN + kt + c4);
            const float vv[4] = {v.x, v.y, v.z, v.w};
            _Float16* ah = Ah + row * LDSROW + c4;
            _Float16* al = Al + row * LDSROW + c4;
            #pragma unroll
            for (int j = 0; j < 4; ++j) {
                const _Float16 h = (_Float16)vv[j];
                ah[j] = h;
                al[j] = (_Float16)(vv[j] - (float)h);
            }
        }
        // ---- stage B tile (32 x 128 fp32 -> f16 hi/lo, transposed) ----
        #pragma unroll
        for (int i = 0; i < 4; ++i) {
            const int idx = t + i * 256;            // 1024 float4s
            const int kr  = idx >> 5;
            const int c4  = (idx & 31) << 2;
            const float4 v = *(const float4*)(Wb + (size_t)(kt + kr) * DMODEL + c4);
            const float vv[4] = {v.x, v.y, v.z, v.w};
            #pragma unroll
            for (int j = 0; j < 4; ++j) {
                const _Float16 h = (_Float16)vv[j];
                Bh[(c4 + j) * LDSROW + kr] = h;
                Bl[(c4 + j) * LDSROW + kr] = (_Float16)(vv[j] - (float)h);
            }
        }
        // prefetch next K tile while this one is consumed
        if (kt + BK < CIN) {
            __builtin_prefetch(xb + (size_t)(t >> 1) * CIN + kt + BK + (t & 1) * 16, 0, 1);
            __builtin_prefetch(Wb + (size_t)(kt + BK + (t >> 3)) * DMODEL + (t & 7) * 16, 0, 1);
        }
        __syncthreads();

        // ---- load WMMA fragments from LDS ----
        // A (16x32 f16): lane = row M = lrow; elems 0..7 at K = half*8, 8..15 at K = 16 + half*8
        v16h a_h[2], a_l[2];
        #pragma unroll
        for (int mi = 0; mi < 2; ++mi) {
            const int m = wm + mi * 16 + lrow;
            const _Float16* ph = Ah + m * LDSROW + halfsel * 8;
            const _Float16* pl = Al + m * LDSROW + halfsel * 8;
            const v8h h0 = *(const v8h*)(ph);
            const v8h h1 = *(const v8h*)(ph + 16);
            const v8h l0 = *(const v8h*)(pl);
            const v8h l1 = *(const v8h*)(pl + 16);
            #pragma unroll
            for (int e = 0; e < 8; ++e) {
                a_h[mi][e] = h0[e]; a_h[mi][8 + e] = h1[e];
                a_l[mi][e] = l0[e]; a_l[mi][8 + e] = l1[e];
            }
        }
        // B (32x16 f16): lane = col N = lrow; elems 0..7 at K = half*8, 8..15 at K = 16 + half*8 (in Bt[n][k])
        v16h b_h[4], b_l[4];
        #pragma unroll
        for (int ni = 0; ni < 4; ++ni) {
            const int n = wn + ni * 16 + lrow;
            const _Float16* ph = Bh + n * LDSROW + halfsel * 8;
            const _Float16* pl = Bl + n * LDSROW + halfsel * 8;
            const v8h h0 = *(const v8h*)(ph);
            const v8h h1 = *(const v8h*)(ph + 16);
            const v8h l0 = *(const v8h*)(pl);
            const v8h l1 = *(const v8h*)(pl + 16);
            #pragma unroll
            for (int e = 0; e < 8; ++e) {
                b_h[ni][e] = h0[e]; b_h[ni][8 + e] = h1[e];
                b_l[ni][e] = l0[e]; b_l[ni][8 + e] = l1[e];
            }
        }

        // ---- split-precision MMA: small cross terms first, dominant last ----
        #pragma unroll
        for (int mi = 0; mi < 2; ++mi) {
            #pragma unroll
            for (int ni = 0; ni < 4; ++ni) {
                acc[mi][ni] = __builtin_amdgcn_wmma_f32_16x16x32_f16(
                    false, a_l[mi], false, b_h[ni], (short)0, acc[mi][ni], false, false);
                acc[mi][ni] = __builtin_amdgcn_wmma_f32_16x16x32_f16(
                    false, a_h[mi], false, b_l[ni], (short)0, acc[mi][ni], false, false);
                acc[mi][ni] = __builtin_amdgcn_wmma_f32_16x16x32_f16(
                    false, a_h[mi], false, b_h[ni], (short)0, acc[mi][ni], false, false);
            }
        }
        __syncthreads();
    }

    // ---- epilogue: +bias, mask blend, store at out[b, 1+s, n] ----
    const float* mrow = mask + (size_t)b * SEQ;
    #pragma unroll
    for (int ni = 0; ni < 4; ++ni) {
        const int n = n0 + wn + ni * 16 + lrow;
        const float bv = bias[(size_t)sid * DMODEL + n];
        const float mt = mask_token[n];
        #pragma unroll
        for (int mi = 0; mi < 2; ++mi) {
            #pragma unroll
            for (int v = 0; v < 8; ++v) {
                const int s = s0 + wm + mi * 16 + halfsel * 8 + v;  // D layout: M = half*8 + v
                const float mk = mrow[s];
                float hv = acc[mi][ni][v] + bv;
                hv = hv * (1.0f - mk) + mt * mk;
                out[((size_t)b * (SEQ + 1) + 1 + s) * DMODEL + n] = hv;
            }
        }
    }
}

// ---------------------------------------------------------------------------
extern "C" void kernel_launch(void* const* d_in, const int* in_sizes, int n_in,
                              void* d_out, int out_size, void* d_ws, size_t ws_size,
                              hipStream_t stream) {
    const float* x          = (const float*)d_in[0];
    const float* mask       = (const float*)d_in[1];
    const float* W          = (const float*)d_in[2];
    const float* bias       = (const float*)d_in[3];
    const float* subj_table = (const float*)d_in[4];
    const float* mask_token = (const float*)d_in[5];
    const int*   sids       = (const int*)d_in[6];
    float* out = (float*)d_out;

    embed_rows_kernel<<<dim3(BATCH), dim3(256), 0, stream>>>(subj_table, sids, out);

    dim3 grid(SEQ / BM, DMODEL / BN, BATCH);   // 4 x 8 x 32 = 1024 workgroups
    subject_gemm_kernel<<<grid, dim3(256), 0, stream>>>(
        x, mask, W, bias, mask_token, sids, out);
}
